// DscaGRUCell_19172734009720
// MI455X (gfx1250) — compile-verified, matrix operand order checked by reading, not run on hardware
//
#include <hip/hip_runtime.h>
#include <hip/hip_bf16.h>
#include <math.h>

// ---------------------------------------------------------------------------
// CDNA5 (gfx1250) WMMA GEMM pipeline for the DscaGRU cell.
// v_wmma_f32_16x16x32_bf16 core, 64-wide K slabs (16 WMMA per barrier pair),
// global_load_b128 staging, v_perm_b32 bf16 packing, ds_store_b128 into LDS,
// contiguous ds_load_b128 fragment loads (ISA 7.12.2 16-bit operand layout).
// ---------------------------------------------------------------------------

typedef __attribute__((ext_vector_type(16))) __bf16 v16bf;
typedef __attribute__((ext_vector_type(8)))  __bf16 v8bf;
typedef __attribute__((ext_vector_type(8)))  float  v8f;

#define TILE_M 128
#define TILE_N 128
#define TILE_K 64   // staged K slab; two 32-wide WMMA k-steps per slab

// Round two fp32 to bf16 (RNE) and pack into one dword with v_perm_b32.
__device__ __forceinline__ unsigned pack_bf16(float a, float b) {
  unsigned ua = __float_as_uint(a);
  ua += 0x7FFFu + ((ua >> 16) & 1u);
  unsigned ub = __float_as_uint(b);
  ub += 0x7FFFu + ((ub >> 16) & 1u);
  // bytes 0..3 from src1(ua), 4..7 from src0(ub): take hi16 of each
  return __builtin_amdgcn_perm(ub, ua, 0x07060302u);
}

// Load one 16x32 bf16 WMMA operand fragment from LDS.
// Lane L (row/col = L&15) holds K = kb..kb+7 and kb+16..kb+23,
// kb = (L<16) ? 0 : 8  -> two contiguous 16B LDS loads.
__device__ __forceinline__ v16bf ld_frag(const __bf16* p) {
  v8bf lo = *(const v8bf*)p;
  v8bf hi = *(const v8bf*)(p + 16);
  v16bf r;
#pragma unroll
  for (int i = 0; i < 8; ++i) { r[i] = lo[i]; r[i + 8] = hi[i]; }
  return r;
}

// Stage a 128x64 A slab and a 64x128 B slab (stored [n][k]) into LDS.
// GUARD=false: straight-line b128 loads + b128 stores, no branches.
// GUARD=true:  K-tail slab with bounds checks and zero fill.
template <bool GUARD>
__device__ __forceinline__ void stage_tiles(
    const float* __restrict__ A, int lda,
    const float* __restrict__ W, int N, int K, int k0,
    int blockRow, int blockCol, int t,
    __bf16* sA, __bf16* sB, bool prefetchNext)
{
  // ---- A slab: 128 rows x 64 k = 1024 8-float chunks, 4 per thread ----
#pragma unroll
  for (int i = 0; i < 4; ++i) {
    int chunk = t + i * 256;          // 0..1023
    int row   = chunk >> 3;           // 0..127
    int c8    = (chunk & 7) << 3;     // 0,8,..,56
    const float* gp = A + (size_t)(blockRow + row) * lda + (k0 + c8);
    float v[8];
    if (!GUARD) {
      float4 a = *(const float4*)gp;
      float4 b = *(const float4*)(gp + 4);
      v[0]=a.x; v[1]=a.y; v[2]=a.z; v[3]=a.w;
      v[4]=b.x; v[5]=b.y; v[6]=b.z; v[7]=b.w;
      if (prefetchNext) __builtin_prefetch(gp + TILE_K, 0, 3);
    } else {
#pragma unroll
      for (int e = 0; e < 8; ++e)
        v[e] = (k0 + c8 + e < K) ? gp[e] : 0.f;
    }
    uint4 p;
    p.x = pack_bf16(v[0], v[1]);
    p.y = pack_bf16(v[2], v[3]);
    p.z = pack_bf16(v[4], v[5]);
    p.w = pack_bf16(v[6], v[7]);
    *(uint4*)(sA + row * TILE_K + c8) = p;
  }

  // ---- B slab: each thread owns an 8(k) x 4(n) micro-tile ----
  {
    int kr = (t >> 5) << 3;           // 0,8,..,56
    int n0 = (t & 31) << 2;           // 0,4,..,124
    float r[8][4];
#pragma unroll
    for (int j = 0; j < 8; ++j) {
      if (!GUARD || (k0 + kr + j) < K) {
        const float* gp = W + (size_t)(k0 + kr + j) * N + (blockCol + n0);
        float4 v = *(const float4*)gp;
        r[j][0] = v.x; r[j][1] = v.y; r[j][2] = v.z; r[j][3] = v.w;
      } else {
        r[j][0] = r[j][1] = r[j][2] = r[j][3] = 0.f;
      }
    }
#pragma unroll
    for (int c = 0; c < 4; ++c) {
      uint4 p;
      p.x = pack_bf16(r[0][c], r[1][c]);
      p.y = pack_bf16(r[2][c], r[3][c]);
      p.z = pack_bf16(r[4][c], r[5][c]);
      p.w = pack_bf16(r[6][c], r[7][c]);
      *(uint4*)(sB + (n0 + c) * TILE_K + kr) = p;
    }
  }
}

// One 32-wide K step: 2 A frags, 4 B frags, 8 WMMAs.
__device__ __forceinline__ void mma_step(
    const __bf16* sA, const __bf16* sB,
    int wm, int wn, int lrow, int kb, int ks, v8f acc[2][4])
{
  v16bf af[2], bfr[4];
#pragma unroll
  for (int mt = 0; mt < 2; ++mt)
    af[mt] = ld_frag(&sA[(wm * 32 + mt * 16 + lrow) * TILE_K + ks * 32 + kb]);
#pragma unroll
  for (int nt = 0; nt < 4; ++nt)
    bfr[nt] = ld_frag(&sB[(wn * 64 + nt * 16 + lrow) * TILE_K + ks * 32 + kb]);
#pragma unroll
  for (int mt = 0; mt < 2; ++mt)
#pragma unroll
    for (int nt = 0; nt < 4; ++nt)
      acc[mt][nt] = __builtin_amdgcn_wmma_f32_16x16x32_bf16(
          false, af[mt], false, bfr[nt], (short)0, acc[mt][nt], false, false);
}

// Epilogue modes:
//   0: C = acc
//   1: C = acc + bias[n]
//   2: C = tanh(acc + bias[n])
//   3: C = addend[m,n] + tanh(acc + bias[n])
__global__ __launch_bounds__(256) void gemm_bf16_wmma(
    const float* __restrict__ A, int lda,
    const float* __restrict__ W,          // K x N, row stride N
    const float* __restrict__ bias,
    const float* __restrict__ addend,
    float* __restrict__ C,
    int K, int N, int epi)
{
  __shared__ __align__(16) __bf16 sA[TILE_M * TILE_K];   // [row][k], 16KB
  __shared__ __align__(16) __bf16 sB[TILE_N * TILE_K];   // [n][k],   16KB

  const int t    = threadIdx.x;
  const int lane = t & 31;
  const int w    = t >> 5;        // 8 waves
  const int wm   = w >> 1;        // 0..3 -> 32-row sub-tile
  const int wn   = w & 1;         // 0..1 -> 64-col sub-tile
  const int lrow = lane & 15;
  const int kb   = (lane & 16) ? 8 : 0;

  const int blockRow = blockIdx.y * TILE_M;   // M = 4096, always divisible
  const int blockCol = blockIdx.x * TILE_N;   // N in {256,512,1536}

  v8f acc[2][4];
#pragma unroll
  for (int mt = 0; mt < 2; ++mt)
#pragma unroll
    for (int nt = 0; nt < 4; ++nt)
      acc[mt][nt] = v8f{};

  const int ktFull   = K >> 6;          // full 64-wide K slabs
  const bool hasTail = (K & 63) != 0;
  const int ktiles   = ktFull + (hasTail ? 1 : 0);

  for (int kt = 0; kt < ktFull; ++kt) {
    stage_tiles<false>(A, lda, W, N, K, kt * TILE_K, blockRow, blockCol, t,
                       sA, sB, kt + 1 < ktiles);
    __syncthreads();
    mma_step(sA, sB, wm, wn, lrow, kb, 0, acc);
    mma_step(sA, sB, wm, wn, lrow, kb, 1, acc);
    __syncthreads();
  }

  if (hasTail) {
    stage_tiles<true>(A, lda, W, N, K, ktFull * TILE_K, blockRow, blockCol, t,
                      sA, sB, false);
    __syncthreads();
    mma_step(sA, sB, wm, wn, lrow, kb, 0, acc);
    mma_step(sA, sB, wm, wn, lrow, kb, 1, acc);
    __syncthreads();
  }

  // ---- epilogue: C/D layout, lane<16 -> rows r, lane>=16 -> rows r+8 ----
#pragma unroll
  for (int mt = 0; mt < 2; ++mt) {
#pragma unroll
    for (int nt = 0; nt < 4; ++nt) {
#pragma unroll
      for (int r = 0; r < 8; ++r) {
        int rr   = (lane & 16) ? (r + 8) : r;
        int grow = blockRow + wm * 32 + mt * 16 + rr;
        int gcol = blockCol + wn * 64 + nt * 16 + (lane & 15);
        float v  = acc[mt][nt][r];
        if (epi == 1)      v += bias[gcol];
        else if (epi == 2) v = tanhf(v + bias[gcol]);
        else if (epi == 3) v = addend[(size_t)grow * N + gcol] + tanhf(v + bias[gcol]);
        C[(size_t)grow * N + gcol] = v;
      }
    }
  }
}

// ---------------------------------------------------------------------------
// Row softmax over 256 elements (one block per batch row, 256 threads).
// ---------------------------------------------------------------------------
__device__ __forceinline__ float block_softmax256(float v, float* red) {
  const int t = threadIdx.x;
  __syncthreads();
  red[t] = v;
  __syncthreads();
#pragma unroll
  for (int s = 128; s > 0; s >>= 1) {
    if (t < s) red[t] = fmaxf(red[t], red[t + s]);
    __syncthreads();
  }
  float mx = red[0];
  __syncthreads();
  float e = __expf(v - mx);
  red[t] = e;
  __syncthreads();
#pragma unroll
  for (int s = 128; s > 0; s >>= 1) {
    if (t < s) red[t] += red[t + s];
    __syncthreads();
  }
  float sum = red[0];
  __syncthreads();
  return e / sum;
}

__global__ __launch_bounds__(256) void att_softmax_kernel(
    const float* __restrict__ emb1, const float* __restrict__ emb2,
    const float* __restrict__ t1, const float* __restrict__ t2,
    const float* __restrict__ t3, const float* __restrict__ s1,
    const float* __restrict__ s2, const float* __restrict__ s3,
    const float* __restrict__ crohis_tm,
    float* __restrict__ x, float* __restrict__ crohis_new)
{
  __shared__ float red[256];
  const int m = blockIdx.x;
  const int t = threadIdx.x;
  const size_t idx = (size_t)m * 256 + t;

  float e1  = emb1[idx], e2 = emb2[idx];
  float v12 = tanhf(t1[idx] + t2[idx]) * t3[idx];
  float v21 = tanhf(s1[idx] + s2[idx]) * s3[idx];

  float p12 = block_softmax256(v12, red);
  float p21 = block_softmax256(v21, red);

  float x1 = e1 * p12;
  float x2 = e2 * p21;
  size_t r0 = (size_t)m * 512 + t;
  x[r0]       = x1;
  x[r0 + 256] = x2;
  crohis_new[r0]       = 0.5f * crohis_tm[r0]       + 0.5f * x1;
  crohis_new[r0 + 256] = 0.5f * crohis_tm[r0 + 256] + 0.5f * x2;
}

// ---------------------------------------------------------------------------
// Fused GRU gates: z, r, hh, h.
// ---------------------------------------------------------------------------
__global__ __launch_bounds__(256) void gru_final_kernel(
    const float* __restrict__ mx, const float* __restrict__ mi,
    const float* __restrict__ hprev, float* __restrict__ h, int total)
{
  int i = blockIdx.x * 256 + threadIdx.x;
  if (i >= total) return;
  int m = i >> 9;        // U = 512
  int u = i & 511;
  const float* mxr = mx + (size_t)m * 1536;
  const float* mir = mi + (size_t)m * 1536;
  float z  = 1.f / (1.f + __expf(-(mxr[u] + mir[u])));
  float r  = 1.f / (1.f + __expf(-(mxr[512 + u] + mir[512 + u])));
  float hp = hprev[i];
  float hh = tanhf(mxr[1024 + u] + r * mir[1024 + u]);
  h[i] = z * hp + (1.f - z) * hh;
}

// ---------------------------------------------------------------------------
// Launcher
// ---------------------------------------------------------------------------
extern "C" void kernel_launch(void* const* d_in, const int* in_sizes, int n_in,
                              void* d_out, int out_size, void* d_ws, size_t ws_size,
                              hipStream_t stream)
{
  const float* inputs    = (const float*)d_in[0];   // 4096 x 4000
  const float* h_tm1     = (const float*)d_in[1];   // 4096 x 512
  const float* crohis_tm = (const float*)d_in[2];   // 4096 x 512
  const float* w_e1      = (const float*)d_in[3];   // 2000 x 256
  const float* w_e2      = (const float*)d_in[4];   // 2000 x 256
  const float* w12w      = (const float*)d_in[5];   // 256 x 256
  const float* w12u      = (const float*)d_in[6];
  const float* w12v      = (const float*)d_in[7];
  const float* w21w      = (const float*)d_in[8];
  const float* w21u      = (const float*)d_in[9];
  const float* w21v      = (const float*)d_in[10];
  const float* cro_w     = (const float*)d_in[11];  // 512 x 512
  const float* cro_b     = (const float*)d_in[12];  // 512
  const float* dec_w     = (const float*)d_in[13];  // 512 x 512
  const float* dec_b     = (const float*)d_in[14];  // 512
  const float* kern      = (const float*)d_in[15];  // 512 x 1536
  const float* rker      = (const float*)d_in[16];  // 512 x 1536
  const float* bias      = (const float*)d_in[17];  // 1536

  const int B = 4096, D = 4000, D1 = 2000, E = 256, M = 512, U = 512, N3 = 1536;

  float* ws    = (float*)d_ws;
  float* emb1  = ws; ws += (size_t)B * E;
  float* emb2  = ws; ws += (size_t)B * E;
  float* t1    = ws; ws += (size_t)B * E;
  float* t2    = ws; ws += (size_t)B * E;
  float* t3    = ws; ws += (size_t)B * E;
  float* s1    = ws; ws += (size_t)B * E;
  float* s2    = ws; ws += (size_t)B * E;
  float* s3    = ws; ws += (size_t)B * E;
  float* xbuf  = ws; ws += (size_t)B * M;
  float* cro   = ws; ws += (size_t)B * M;
  float* hprev = ws; ws += (size_t)B * U;
  float* mi    = ws; ws += (size_t)B * N3;
  float* mx    = t1;   // t1..s3 (6*B*E == B*1536) are dead after att kernel

  float* h_out   = (float*)d_out;            // 4096 x 512
  float* cro_out = h_out + (size_t)B * U;    // 4096 x 512

  dim3 blk(256);
  dim3 gE(E / TILE_N, B / TILE_M);    // (2, 32)
  dim3 gM(M / TILE_N, B / TILE_M);    // (4, 32)
  dim3 g3(N3 / TILE_N, B / TILE_M);   // (12, 32)

  // Embeddings: inputs[:, :2000] @ We1 ; inputs[:, 2000:] @ We2
  gemm_bf16_wmma<<<gE, blk, 0, stream>>>(inputs,       D, w_e1, nullptr, nullptr, emb1, D1, E, 0);
  gemm_bf16_wmma<<<gE, blk, 0, stream>>>(inputs + D1,  D, w_e2, nullptr, nullptr, emb2, D1, E, 0);

  // Attention projections (all 4096x256x256)
  gemm_bf16_wmma<<<gE, blk, 0, stream>>>(emb1, E, w12w, nullptr, nullptr, t1, E, E, 0);
  gemm_bf16_wmma<<<gE, blk, 0, stream>>>(emb2, E, w12u, nullptr, nullptr, t2, E, E, 0);
  gemm_bf16_wmma<<<gE, blk, 0, stream>>>(emb2, E, w12v, nullptr, nullptr, t3, E, E, 0);
  gemm_bf16_wmma<<<gE, blk, 0, stream>>>(emb2, E, w21w, nullptr, nullptr, s1, E, E, 0);
  gemm_bf16_wmma<<<gE, blk, 0, stream>>>(emb1, E, w21u, nullptr, nullptr, s2, E, E, 0);
  gemm_bf16_wmma<<<gE, blk, 0, stream>>>(emb1, E, w21v, nullptr, nullptr, s3, E, E, 0);

  // Softmax attention -> x, plus crohis_new written straight to d_out
  att_softmax_kernel<<<B, blk, 0, stream>>>(emb1, emb2, t1, t2, t3, s1, s2, s3,
                                            crohis_tm, xbuf, cro_out);

  // cro = tanh(crohis_tm @ dec_w + dec_b)
  gemm_bf16_wmma<<<gM, blk, 0, stream>>>(crohis_tm, M, dec_w, dec_b, nullptr, cro, M, M, 2);
  // h_prev = h_tm1 + tanh(cro @ cro_w + cro_b)
  gemm_bf16_wmma<<<gM, blk, 0, stream>>>(cro, M, cro_w, cro_b, h_tm1, hprev, M, U, 3);

  // mx = x @ kernel + bias ; mi = h_prev @ recurrent_kernel
  gemm_bf16_wmma<<<g3, blk, 0, stream>>>(xbuf,  M, kern, bias,    nullptr, mx, M, N3, 1);
  gemm_bf16_wmma<<<g3, blk, 0, stream>>>(hprev, U, rker, nullptr, nullptr, mi, U, N3, 0);

  // Fused GRU gates -> h
  gru_final_kernel<<<(B * U) / 256, blk, 0, stream>>>(mx, mi, hprev, h_out, B * U);
}
